// LHTLayer_12661563589248
// MI455X (gfx1250) — matrix operand order repaired and checked by reading, not compiled, not run on hardware
//
#include <hip/hip_runtime.h>

// ---------- types ----------
typedef __attribute__((ext_vector_type(16))) __bf16   v16bf;
typedef __attribute__((ext_vector_type(8)))  float    v8f;
typedef __attribute__((ext_vector_type(4)))  unsigned v4u;

// sizes
#define BB   2
#define SS   512
#define DD   512
#define HH   8
#define DFF_ 2048
#define DH_  64

__device__ __forceinline__ unsigned short f2bf(float f) {
    unsigned u = __float_as_uint(f);
    unsigned r = u + 0x7FFFu + ((u >> 16) & 1u);   // round-to-nearest-even
    return (unsigned short)(r >> 16);
}

// ---------------------------------------------------------------------------
// Generic bf16 WMMA GEMM:  C[M,N] = A[M,K] * BT[N,K]^T  (+bias) (+epilogue)
// 256 threads = 8 waves, templated on wave arrangement:
//   WN==2: block tile 64(M) x 128(N), waves 4x2
//   WN==1: block tile 128(M) x 64(N), waves 8x1      (for N==64 GEMMs)
// All launches have M % blockM == 0 and N % blockN == 0 -> no guards.
// K-slices are double-buffered in LDS: async-to-LDS loads (ASYNCcnt) for the
// next K-step are issued before the WMMAs of the current step, hidden behind
// compute; one s_wait_asynccnt + one barrier per step.
// mode 0: Cf = acc + bias            (f32)
// mode 1: Cf = gelu(acc + bias)      (f32)
// mode 2: Cf = acc + bias + res      (f32)
// mode 3: Cb[(m*64 + n%64)*512 + n/64] = bf16(acc + bias)   (A-tensor permute)
// ---------------------------------------------------------------------------
template<int WN>
__global__ __launch_bounds__(256) void gemm_bf16_kernel(
    const unsigned short* __restrict__ A, const unsigned short* __restrict__ BT,
    const float* __restrict__ bias, const float* __restrict__ res,
    float* __restrict__ Cf, unsigned short* __restrict__ Cb,
    int M, int N, int Kd, int lda, int ldb, int ldc,
    long long sA, long long sBT, long long sC, long long sRes, int mode)
{
    constexpr int BM  = (WN == 2) ? 64 : 128;
    constexpr int BN  = (WN == 2) ? 128 : 64;
    constexpr int ACH = BM * 4;   // 16B chunks per A K-slice (256 or 512)
    constexpr int BCH = BN * 4;   // 16B chunks per B K-slice (512 or 256)

    const int bat = blockIdx.z;
    A  += (long long)bat * sA;
    BT += (long long)bat * sBT;

    const int tid  = threadIdx.x;
    const int lane = tid & 31;
    const int wid  = tid >> 5;
    const int wm   = (WN == 2) ? (wid & 3) : wid;
    const int wn   = (WN == 2) ? (wid >> 2) : 0;

    const long long arow0 = (long long)blockIdx.y * BM;   // global M base of block
    const long long brow0 = (long long)blockIdx.x * BN;   // global N base of block
    const int m0 = (int)arow0 + wm * 16;
    const int nb = (int)brow0 + wn * 64;

    const int ahalf = (lane & 16) ? 8  : 0;   // A frag: K sub-block select
    const int bhalf = (lane & 16) ? 8  : 0;   // B frag: K sub-block select
    const int ncol  = lane & 15;

    __shared__ unsigned short As[2][128 * 32];   // double-buffered K-slices
    __shared__ unsigned short Bs[2][128 * 32];
    unsigned asb[2], bsb[2];
    asb[0] = (unsigned)(size_t)(void*)As[0];
    asb[1] = (unsigned)(size_t)(void*)As[1];
    bsb[0] = (unsigned)(size_t)(void*)Bs[0];
    bsb[1] = (unsigned)(size_t)(void*)Bs[1];

    // async staging of one K-slice into buffer `buf` (straight-line, no masks)
    auto stage = [&](int buf, int kk) {
        {
            int c = tid, row = c >> 2, ch = c & 3;
            const unsigned short* gp = A + (arow0 + row) * lda + kk + ch * 8;
            unsigned lofs = asb[buf] + (unsigned)((row * 32 + ch * 8) * 2);
            asm volatile("global_load_async_to_lds_b128 %0, %1, off"
                         :: "v"(lofs), "v"((unsigned long long)(size_t)gp) : "memory");
            if constexpr (ACH == 512) {
                c = tid + 256; row = c >> 2; ch = c & 3;
                gp = A + (arow0 + row) * lda + kk + ch * 8;
                lofs = asb[buf] + (unsigned)((row * 32 + ch * 8) * 2);
                asm volatile("global_load_async_to_lds_b128 %0, %1, off"
                             :: "v"(lofs), "v"((unsigned long long)(size_t)gp) : "memory");
            }
        }
        {
            int c = tid, row = c >> 2, ch = c & 3;
            const unsigned short* gp = BT + (brow0 + row) * ldb + kk + ch * 8;
            unsigned lofs = bsb[buf] + (unsigned)((row * 32 + ch * 8) * 2);
            asm volatile("global_load_async_to_lds_b128 %0, %1, off"
                         :: "v"(lofs), "v"((unsigned long long)(size_t)gp) : "memory");
            if constexpr (BCH == 512) {
                c = tid + 256; row = c >> 2; ch = c & 3;
                gp = BT + (brow0 + row) * ldb + kk + ch * 8;
                lofs = bsb[buf] + (unsigned)((row * 32 + ch * 8) * 2);
                asm volatile("global_load_async_to_lds_b128 %0, %1, off"
                             :: "v"(lofs), "v"((unsigned long long)(size_t)gp) : "memory");
            }
        }
    };

    v8f acc[4];
    for (int t = 0; t < 4; ++t)
        for (int e = 0; e < 8; ++e) acc[t][e] = 0.0f;

    union Frag { v4u u[2]; v16bf v; };

    // prologue: stage K-slice 0
    stage(0, 0);
    asm volatile("s_wait_asynccnt 0x0" ::: "memory");
    __syncthreads();

    int cur = 0;
    for (int kk = 0; kk < Kd; kk += 32) {
        const int nxt = cur ^ 1;
        if (kk + 32 < Kd) stage(nxt, kk + 32);   // prefetch next slice (async)

        // --- fragments from current LDS buffer ---
        Frag af;
        {
            const unsigned short* ap =
                As[cur] + (wm * 16 + (lane & 15)) * 32 + ahalf;
            af.u[0] = *(const v4u*)(ap);
            af.u[1] = *(const v4u*)(ap + 16);
        }
        Frag bfr[4];
        #pragma unroll
        for (int t = 0; t < 4; ++t) {
            const unsigned short* bp =
                Bs[cur] + (wn * 64 + t * 16 + ncol) * 32 + bhalf;
            bfr[t].u[0] = *(const v4u*)(bp);
            bfr[t].u[1] = *(const v4u*)(bp + 16);
        }
        #pragma unroll
        for (int t = 0; t < 4; ++t)
            acc[t] = __builtin_amdgcn_wmma_f32_16x16x32_bf16(
                         false, af.v, false, bfr[t].v, (short)0, acc[t], false, false);

        asm volatile("s_wait_asynccnt 0x0" ::: "memory");  // next slice landed
        __syncthreads();                                   // all waves done with cur
        cur = nxt;
    }

    const int cm0 = m0 + ((lane & 16) ? 8 : 0);
    const long long cbase = (long long)bat * sC;
    for (int t = 0; t < 4; ++t) {
        int n = nb + t * 16 + ncol;
        float bv = bias ? bias[n] : 0.0f;
        #pragma unroll
        for (int r = 0; r < 8; ++r) {
            int m = cm0 + r;
            float v = acc[t][r] + bv;
            if (mode == 0) {
                Cf[cbase + (long long)m * ldc + n] = v;
            } else if (mode == 1) {
                Cf[cbase + (long long)m * ldc + n] =
                    0.5f * v * (1.0f + erff(v * 0.70710678118654752f));
            } else if (mode == 2) {
                Cf[cbase + (long long)m * ldc + n] =
                    v + res[(long long)bat * sRes + (long long)m * ldc + n];
            } else { // mode 3: A = (h@Wc2).reshape(b,s,d,K,K) -> store [b,j,kl,d] bf16
                Cb[((long long)m * 64 + (n & 63)) * 512 + (n >> 6)] = f2bf(v);
            }
        }
    }
}

// ---------------------------------------------------------------------------
// elementwise helpers
// ---------------------------------------------------------------------------
__global__ void convert_bf16_kernel(const float* __restrict__ in,
                                    unsigned short* __restrict__ out, long long n) {
    long long i = (long long)blockIdx.x * blockDim.x + threadIdx.x;
    if (i < n) out[i] = f2bf(in[i]);
}

__global__ void transpose_bf16_kernel(const float* __restrict__ in,
                                      unsigned short* __restrict__ out, int R, int C) {
    long long i = (long long)blockIdx.x * blockDim.x + threadIdx.x;
    long long total = (long long)R * C;
    if (i >= total) return;
    int r = (int)(i / C), c = (int)(i % C);
    out[(long long)c * R + r] = f2bf(in[i]);
}

// q/k f32 [b,s,(h,dh)] -> bf16 [b,h,s,dh]
__global__ void pack_qk_kernel(const float* __restrict__ in,
                               unsigned short* __restrict__ out) {
    long long i = (long long)blockIdx.x * blockDim.x + threadIdx.x;
    if (i >= (long long)BB * SS * DD) return;
    int b = (int)(i >> 18);
    int rem = (int)(i & 262143);
    int s = rem >> 9, col = rem & 511;
    int h = col >> 6, dh = col & 63;
    out[(((long long)b * HH + h) * SS + s) * DH_ + dh] = f2bf(in[i]);
}

// v f32 [b,s,(h,dh)] -> bf16 [b,h,dh,s]  (acts as B^T for ctx GEMM)
__global__ void pack_v_kernel(const float* __restrict__ in,
                              unsigned short* __restrict__ out) {
    long long i = (long long)blockIdx.x * blockDim.x + threadIdx.x;
    if (i >= (long long)BB * SS * DD) return;
    int b = (int)(i >> 18);
    int rem = (int)(i & 262143);
    int s = rem >> 9, col = rem & 511;
    int h = col >> 6, dh = col & 63;
    out[(((long long)b * HH + h) * DH_ + dh) * SS + s] = f2bf(in[i]);
}

// ctx f32 [b,h,s,dh] -> bf16 [(b,s),(h,dh)]
__global__ void pack_ctx_kernel(const float* __restrict__ in,
                                unsigned short* __restrict__ out) {
    long long i = (long long)blockIdx.x * blockDim.x + threadIdx.x;
    if (i >= (long long)BB * SS * DD) return;
    int b = (int)(i >> 18);
    int rem = (int)(i & 262143);
    int h = rem >> 15;
    int rem2 = rem & 32767;
    int s = rem2 >> 6, dh = rem2 & 63;
    out[((long long)b * SS + s) * DD + h * DH_ + dh] = f2bf(in[i]);
}

// softmax over rows of 512 with 1/sqrt(64) scaling; writes f32 (in place) + bf16
__global__ __launch_bounds__(256) void softmax_kernel(float* __restrict__ sc,
                                                      unsigned short* __restrict__ abf) {
    const long long base = (long long)blockIdx.x * 512;
    const int t = threadIdx.x;
    __shared__ float red[256];
    float v0 = sc[base + t] * 0.125f;
    float v1 = sc[base + t + 256] * 0.125f;
    float m = fmaxf(v0, v1);
    red[t] = m; __syncthreads();
    for (int s = 128; s > 0; s >>= 1) {
        if (t < s) red[t] = fmaxf(red[t], red[t + s]);
        __syncthreads();
    }
    m = red[0]; __syncthreads();
    float e0 = __expf(v0 - m), e1 = __expf(v1 - m);
    red[t] = e0 + e1; __syncthreads();
    for (int s = 128; s > 0; s >>= 1) {
        if (t < s) red[t] += red[t + s];
        __syncthreads();
    }
    float inv = 1.0f / red[0];
    float a0 = e0 * inv, a1 = e1 * inv;
    sc[base + t] = a0;        sc[base + t + 256] = a1;
    abf[base + t] = f2bf(a0); abf[base + t + 256] = f2bf(a1);
}

// attn_avg[b,i,j] = mean_h attn[b,h,i,j]
__global__ void attn_avg_kernel(const float* __restrict__ attn,
                                float* __restrict__ avg) {
    long long i = (long long)blockIdx.x * blockDim.x + threadIdx.x;
    if (i >= (long long)BB * SS * SS) return;
    int b = (int)(i >> 18);
    int rem = (int)(i & 262143);
    int ii = rem >> 9, jj = rem & 511;
    float s = 0.0f;
    for (int h = 0; h < HH; ++h)
        s += attn[((((long long)b * HH + h) * SS + ii) * SS) + jj];
    avg[i] = s * 0.125f;
}

// layernorm over rows of 2048 -> bf16
__global__ __launch_bounds__(256) void layernorm_kernel(
    const float* __restrict__ h, const float* __restrict__ g,
    const float* __restrict__ bta, unsigned short* __restrict__ hn) {
    const long long base = (long long)blockIdx.x * DFF_;
    const int t = threadIdx.x;
    __shared__ float rs[256], rs2[256];
    float vals[8];
    float s = 0.0f, s2 = 0.0f;
    for (int e = 0; e < 8; ++e) {
        float v = h[base + t + e * 256];
        vals[e] = v; s += v; s2 += v * v;
    }
    rs[t] = s; rs2[t] = s2; __syncthreads();
    for (int st = 128; st > 0; st >>= 1) {
        if (t < st) { rs[t] += rs[t + st]; rs2[t] += rs2[t + st]; }
        __syncthreads();
    }
    float mean = rs[0] * (1.0f / DFF_);
    float var  = rs2[0] * (1.0f / DFF_) - mean * mean;
    float inv  = rsqrtf(var + 1e-5f);
    for (int e = 0; e < 8; ++e) {
        int c = t + e * 256;
        hn[base + c] = f2bf((vals[e] - mean) * inv * g[c] + bta[c]);
    }
}

// ---------------------------------------------------------------------------
// Fused parallel-transport kernel: per (b,i) over all j:
//   L = T[b,j,j,:] - T[b,i,j,:]            (= -lie, 8x8)
//   G = Taylor6(exp(L))
//   u_t[b,i,k,c] += attn_avg[b,i,j] * sum_l G[k,l]*u_proj[b,j,l*64+c]
// Never materializes lie/Gamma [B,S,S,8,8] (saves ~500MB of HBM traffic);
// T rows and u_proj stay hot in the 192MB L2.
// ---------------------------------------------------------------------------
__global__ __launch_bounds__(256) void fused_transport_kernel(
    const float* __restrict__ T, const float* __restrict__ uproj,
    const float* __restrict__ aavg, unsigned short* __restrict__ ut_bf) {
    const int i = blockIdx.x;     // 0..511
    const int b = blockIdx.y;     // 0..1
    const int t = threadIdx.x;
    __shared__ float Lm[64], Pa[64], Pb[64], G[64], ups[512];

    const float* Tb = T + (long long)b * SS * 32768;
    const float* wrow = aavg + ((long long)b * SS + i) * SS;

    const int k0 = t >> 6,         c0 = t & 63;
    const int k1 = (t + 256) >> 6, c1 = t & 63;
    float acc0 = 0.0f, acc1 = 0.0f;

    for (int j = 0; j < SS; ++j) {
        if (t < 64) {
            float tij = Tb[(long long)i * 32768 + j * 64 + t];
            float tjj = Tb[(long long)j * 32768 + j * 64 + t];
            Lm[t] = tjj - tij;                         // = -lie[b,i,j]
            float Iv = ((t >> 3) == (t & 7)) ? 1.0f : 0.0f;
            Pa[t] = Iv;
            G[t]  = Iv;
        }
        ups[t]       = uproj[((long long)b * SS + j) * DD + t];
        ups[t + 256] = uproj[((long long)b * SS + j) * DD + t + 256];
        __syncthreads();

        float* Pin = Pa; float* Pout = Pb;
        for (int n = 1; n <= 6; ++n) {
            if (t < 64) {
                int k = t >> 3, l = t & 7;
                float s = 0.0f;
                #pragma unroll
                for (int m = 0; m < 8; ++m) s += Pin[k * 8 + m] * Lm[m * 8 + l];
                s *= (1.0f / (float)n);
                Pout[t] = s;
                G[t]   += s;
            }
            __syncthreads();
            float* tmp = Pin; Pin = Pout; Pout = tmp;
        }

        float wij = wrow[j];
        float s0 = 0.0f, s1 = 0.0f;
        #pragma unroll
        for (int l = 0; l < 8; ++l) {
            s0 += G[k0 * 8 + l] * ups[l * 64 + c0];
            s1 += G[k1 * 8 + l] * ups[l * 64 + c1];
        }
        acc0 += wij * s0;
        acc1 += wij * s1;
        __syncthreads();   // before next j overwrites Lm/ups/G
    }
    long long ob = ((long long)b * SS + i) * DD;
    ut_bf[ob + t]       = f2bf(acc0);
    ut_bf[ob + t + 256] = f2bf(acc1);
}

// ---------------------------------------------------------------------------
// host orchestration
// ---------------------------------------------------------------------------
extern "C" void kernel_launch(void* const* d_in, const int* in_sizes, int n_in,
                              void* d_out, int out_size, void* d_ws, size_t ws_size,
                              hipStream_t stream) {
    (void)in_sizes; (void)n_in; (void)out_size; (void)ws_size;
    const float* x   = (const float*)d_in[0];
    const float* u   = (const float*)d_in[1];
    const float* Wq  = (const float*)d_in[2];  const float* bq  = (const float*)d_in[3];
    const float* Wk  = (const float*)d_in[4];  const float* bk  = (const float*)d_in[5];
    const float* Wv  = (const float*)d_in[6];  const float* bv  = (const float*)d_in[7];
    const float* Wo  = (const float*)d_in[8];  const float* bo  = (const float*)d_in[9];
    const float* Wfp = (const float*)d_in[10]; const float* bfp = (const float*)d_in[11];
    const float* Wfo = (const float*)d_in[12]; const float* bfo = (const float*)d_in[13];
    const float* Wc1 = (const float*)d_in[14]; const float* bc1 = (const float*)d_in[15];
    const float* g_ln = (const float*)d_in[16]; const float* b_ln = (const float*)d_in[17];
    const float* Wc2 = (const float*)d_in[18]; const float* bc2 = (const float*)d_in[19];

    float* out_x = (float*)d_out;                    // [B,S,D]
    float* out_u = out_x + (long long)BB * SS * DD;  // [B,S,DF]

    // workspace carve-out (all 256B aligned)
    char* p = (char*)d_ws;
    auto alloc = [&](size_t bytes) {
        char* r = p; p += (bytes + 255) & ~(size_t)255; return r;
    };
    unsigned short* Wc2T = (unsigned short*)alloc(134217728); // [32768,2048] bf16
    float*          T    = (float*)Wc2T;   // ALIAS: Wc2T dead after A-GEMM; T is [2,512,32768] f32
    unsigned short* Abuf = (unsigned short*)alloc(67108864);  // [b,j,kl,d] bf16
    unsigned short* Wc1T = (unsigned short*)alloc(2097152);   // [2048,512]
    unsigned short* WqT  = (unsigned short*)alloc(524288);
    unsigned short* WkT  = (unsigned short*)alloc(524288);
    unsigned short* WvT  = (unsigned short*)alloc(524288);
    unsigned short* WoT  = (unsigned short*)alloc(524288);
    unsigned short* WfpT = (unsigned short*)alloc(65536);     // [512,64]
    unsigned short* WfoT = (unsigned short*)alloc(65536);     // [64,512]
    unsigned short* x_bf = (unsigned short*)alloc(1048576);   // [1024,512]
    unsigned short* u_bf = (unsigned short*)alloc(131072);    // [1024,64]
    float*          hbuf = (float*)alloc(8388608);            // [1024,2048]
    unsigned short* hn   = (unsigned short*)alloc(4194304);   // [1024,2048]
    float*          qf   = (float*)alloc(2097152);
    float*          kf   = (float*)alloc(2097152);
    float*          vf   = (float*)alloc(2097152);
    unsigned short* qbf  = (unsigned short*)alloc(1048576);   // [b,h,s,dh]
    unsigned short* kbf  = (unsigned short*)alloc(1048576);   // [b,h,s,dh]
    unsigned short* vTbf = (unsigned short*)alloc(1048576);   // [b,h,dh,s]
    float*          scf  = (float*)alloc(16777216);           // [b,h,s,s]
    unsigned short* abf  = (unsigned short*)alloc(8388608);   // attn bf16
    float*          aav  = (float*)alloc(2097152);            // [b,s,s]
    float*          ctxf = (float*)alloc(2097152);            // [b,h,s,dh]
    unsigned short* ctxb = (unsigned short*)alloc(1048576);   // [(b,s),(h,dh)]
    float*          upf  = (float*)alloc(2097152);            // [1024,512]
    unsigned short* utb  = (unsigned short*)alloc(1048576);   // [1024,512]

    auto ew = [&](long long n) { return dim3((unsigned)((n + 255) / 256)); };
    auto gemm = [&](const unsigned short* A, const unsigned short* BT,
                    const float* bias, const float* res, float* Cf, unsigned short* Cb,
                    int M, int N, int K, int lda, int ldb, int ldc,
                    long long sA, long long sBT, long long sC, long long sRes,
                    int mode, int batch, int wavesn) {
        int bm = (wavesn == 2) ? 64 : 128;
        int bn = (wavesn == 2) ? 128 : 64;
        dim3 grid((N + bn - 1) / bn, (M + bm - 1) / bm, batch);
        if (wavesn == 2)
            gemm_bf16_kernel<2><<<grid, 256, 0, stream>>>(A, BT, bias, res, Cf, Cb,
                M, N, K, lda, ldb, ldc, sA, sBT, sC, sRes, mode);
        else
            gemm_bf16_kernel<1><<<grid, 256, 0, stream>>>(A, BT, bias, res, Cf, Cb,
                M, N, K, lda, ldb, ldc, sA, sBT, sC, sRes, mode);
    };

    // 1) weight transposes -> bf16 B^T layouts
    transpose_bf16_kernel<<<ew(1048576), 256, 0, stream>>>(Wc1, Wc1T, 512, 2048);
    transpose_bf16_kernel<<<ew(67108864), 256, 0, stream>>>(Wc2, Wc2T, 2048, 32768);
    transpose_bf16_kernel<<<ew(262144), 256, 0, stream>>>(Wq, WqT, 512, 512);
    transpose_bf16_kernel<<<ew(262144), 256, 0, stream>>>(Wk, WkT, 512, 512);
    transpose_bf16_kernel<<<ew(262144), 256, 0, stream>>>(Wv, WvT, 512, 512);
    transpose_bf16_kernel<<<ew(262144), 256, 0, stream>>>(Wo, WoT, 512, 512);
    transpose_bf16_kernel<<<ew(32768), 256, 0, stream>>>(Wfp, WfpT, 64, 512);
    transpose_bf16_kernel<<<ew(32768), 256, 0, stream>>>(Wfo, WfoT, 512, 64);
    convert_bf16_kernel<<<ew(524288), 256, 0, stream>>>(x, x_bf, 524288);
    convert_bf16_kernel<<<ew(65536), 256, 0, stream>>>(u, u_bf, 65536);

    // 2) connection network: h = gelu(x@Wc1+bc1); hn = LN(h)
    gemm(x_bf, Wc1T, bc1, nullptr, hbuf, nullptr,
         1024, 2048, 512, 512, 512, 2048, 0, 0, 0, 0, /*gelu*/1, 1, 2);
    layernorm_kernel<<<1024, 256, 0, stream>>>(hbuf, g_ln, b_ln, hn);

    // 3) A = hn@Wc2+bc2, written permuted bf16 as [b,j,kl,d]
    gemm(hn, Wc2T, bc2, nullptr, nullptr, Abuf,
         1024, 32768, 2048, 2048, 2048, 0, 0, 0, 0, 0, /*permute*/3, 1, 2);

    // 4) T[b,i,(j,kl)] = sum_d x[b,i,d]*A[b,j,d,kl]   (batched over b; aliases Wc2T)
    gemm(x_bf, Abuf, nullptr, nullptr, T, nullptr,
         512, 32768, 512, 512, 512, 32768,
         262144LL, 16777216LL, 16777216LL, 0, 0, 2, 2);

    // 5) attention q/k/v projections
    gemm(x_bf, WqT, bq, nullptr, qf, nullptr, 1024, 512, 512, 512, 512, 512, 0,0,0,0, 0, 1, 2);
    gemm(x_bf, WkT, bk, nullptr, kf, nullptr, 1024, 512, 512, 512, 512, 512, 0,0,0,0, 0, 1, 2);
    gemm(x_bf, WvT, bv, nullptr, vf, nullptr, 1024, 512, 512, 512, 512, 512, 0,0,0,0, 0, 1, 2);
    pack_qk_kernel<<<ew(524288), 256, 0, stream>>>(qf, qbf);
    pack_qk_kernel<<<ew(524288), 256, 0, stream>>>(kf, kbf);
    pack_v_kernel<<<ew(524288), 256, 0, stream>>>(vf, vTbf);

    // 6) scores = q@k^T per (b,h); softmax(+1/8 scale); head-average
    gemm(qbf, kbf, nullptr, nullptr, scf, nullptr,
         512, 512, 64, 64, 64, 512, 32768LL, 32768LL, 262144LL, 0, 0, 16, 2);
    softmax_kernel<<<8192, 256, 0, stream>>>(scf, abf);
    attn_avg_kernel<<<ew(524288), 256, 0, stream>>>(scf, aav);

    // 7) ctx = attn@v per (b,h); x_out = x + ctx@Wo + bo
    gemm(abf, vTbf, nullptr, nullptr, ctxf, nullptr,
         512, 64, 512, 512, 512, 64, 262144LL, 32768LL, 32768LL, 0, 0, 16, 1);
    pack_ctx_kernel<<<ew(524288), 256, 0, stream>>>(ctxf, ctxb);
    gemm(ctxb, WoT, bo, x, out_x, nullptr,
         1024, 512, 512, 512, 512, 512, 0, 0, 0, 0, /*residual*/2, 1, 2);

    // 8) fiber transport: u_proj = u@Wfp+bfp; fused matexp+transport; u_out
    gemm(u_bf, WfpT, bfp, nullptr, upf, nullptr,
         1024, 512, 64, 64, 64, 512, 0, 0, 0, 0, 0, 1, 2);
    fused_transport_kernel<<<dim3(512, 2), 256, 0, stream>>>(T, upf, aav, utb);
    gemm(utb, WfoT, bfo, u, out_u, nullptr,
         1024, 64, 512, 512, 512, 64, 0, 0, 0, 0, /*residual*/2, 1, 1);
}
